// TransformerBlock_17102559773004
// MI455X (gfx1250) — matrix operand order repaired and checked by reading, not compiled, not run on hardware
//
#include <hip/hip_runtime.h>
#include <math.h>

// Problem constants (from reference): B=2, S=2048, H=1024, NH=16, HD=64, DFF=4096
#define Bsz   2
#define Sq    2048
#define Hd    1024
#define NHn   16
#define HDd   64
#define DFFn  4096
#define Mrows (Bsz*Sq)      // 4096 token rows
#define H3    (3*Hd)        // 3072

typedef __attribute__((ext_vector_type(16))) __bf16 v16bf;
typedef __attribute__((ext_vector_type(8)))  float  v8f;
typedef __attribute__((ext_vector_type(4)))  unsigned int v4u;
typedef __attribute__((ext_vector_type(8)))  int v8i;
typedef __attribute__((ext_vector_type(4)))  int v4i;

#if defined(__gfx1250__) && __has_builtin(__builtin_amdgcn_tensor_load_to_lds) && \
    __has_builtin(__builtin_amdgcn_s_wait_tensorcnt)
#define USE_TDM 1
#else
#define USE_TDM 0
#endif

union Frag { v16bf v; unsigned int u[8]; };
union U4   { uint4 q; unsigned short h[8]; };

__device__ __forceinline__ unsigned short f2bf(float f) {
  unsigned int u = __float_as_uint(f);
  u += 0x7FFFu + ((u >> 16) & 1u);        // round-to-nearest-even
  return (unsigned short)(u >> 16);
}

// LDS byte offset of a __shared__ object: generic LDS address maps addr[31:0]
// to the LDS offset (ISA 10.2 aperture rules), so truncation is the offset.
__device__ __forceinline__ unsigned ldsoff(const void* p) {
  return (unsigned)(size_t)p;
}

#if USE_TDM
// Tensor Data Mover: DMA a 2-D tile (tile_rows x tile_k bf16 elements, row
// stride row_stride_elems) from global memory into LDS at lds_off, laid out
// contiguously [tile_rows][tile_k].  D# built per CDNA5 ISA sec. 8.3/8.4.
// Issue once per wave (EXEC ignored); tracked by TENSORcnt.
// This toolchain uses the 6-arg builtin form: (v4u, v8i, v4i, v4i, v8i, i32).
__device__ __forceinline__ void tdm_load_2d(const unsigned short* gptr,
                                            unsigned lds_off,
                                            unsigned tile_k, unsigned tile_rows,
                                            unsigned row_stride_elems) {
  const unsigned long long ga = (unsigned long long)(size_t)gptr;
  v4u g0; v8i g1; v4i g2; v4i g3; v8i g4;
  // group0: [1:0]=count=1 | [63:32]=lds_addr | [120:64]=global_addr | [127:126]=type=2
  g0[0] = 1u;
  g0[1] = lds_off;
  g0[2] = (unsigned)ga;
  g0[3] = (unsigned)(ga >> 32) | 0x80000000u;
  // group1: data_size=1 (2B); tensor_dim0=row_stride; tensor_dim1=huge;
  // tile_dim0=tile_k; tile_dim1=tile_rows; tensor_dim0_stride=row_stride.
  const unsigned td0 = row_stride_elems;
  const unsigned td1 = 0x7fffffffu;
  g1[0] = 0x00010000;                                          // data_size=2B
  g1[1] = (int)((td0 & 0xffffu) << 16);                        // dim0[15:0]
  g1[2] = (int)((td0 >> 16) | ((td1 & 0xffffu) << 16));        // dim0[31:16]|dim1[15:0]
  g1[3] = (int)((td1 >> 16) | (tile_k << 16));                 // dim1[31:16]|tile_dim0
  g1[4] = (int)tile_rows;                                      // tile_dim1 (tile_dim2=0)
  g1[5] = (int)row_stride_elems;                               // dim0_stride[31:0]
  g1[6] = 0;                                                   // dim0_stride[47:32]
  g1[7] = 0;
  g2[0] = 0; g2[1] = 0; g2[2] = 0; g2[3] = 0;                  // 2-D: dims/tiles unused
  g3[0] = 0; g3[1] = 0; g3[2] = 0; g3[3] = 0;
  g4[0] = 0; g4[1] = 0; g4[2] = 0; g4[3] = 0;
  g4[4] = 0; g4[5] = 0; g4[6] = 0; g4[7] = 0;
  __builtin_amdgcn_tensor_load_to_lds(g0, g1, g2, g3, g4, 0);
}
#endif

// ---- WMMA fragment loads from LDS, per CDNA5 ISA 7.12.2 layouts (wave32) ----
__device__ __forceinline__ void load_afrag(Frag& f, const unsigned short* lds, int strideK) {
  const int lane = threadIdx.x & 31;
  const int row  = lane & 15;
  const int kb   = (lane < 16) ? 0 : 8;
  const unsigned int* p = (const unsigned int*)(lds + row * strideK);
#pragma unroll
  for (int j = 0; j < 4; ++j) f.u[j]     = p[(kb >> 1) + j];
#pragma unroll
  for (int j = 0; j < 4; ++j) f.u[4 + j] = p[((kb + 16) >> 1) + j];
}
__device__ __forceinline__ void load_bfrag(Frag& f, const unsigned short* lds, int strideK) {
  const int lane = threadIdx.x & 31;
  const int col  = lane & 15;
  const int kb   = (lane < 16) ? 0 : 8;
  const unsigned int* p = (const unsigned int*)(lds + col * strideK);
#pragma unroll
  for (int j = 0; j < 4; ++j) f.u[j]     = p[(kb >> 1) + j];
#pragma unroll
  for (int j = 0; j < 4; ++j) f.u[4 + j] = p[((kb + 16) >> 1) + j];
}

__device__ __forceinline__ v8f wmma_bf16(const Frag& a, const Frag& b, v8f c) {
  return __builtin_amdgcn_wmma_f32_16x16x32_bf16(false, a.v, false, b.v,
                                                 (short)0, c, false, false);
}

__device__ __forceinline__ float hmax16(float v) {
#pragma unroll
  for (int m = 1; m < 16; m <<= 1) v = fmaxf(v, __shfl_xor(v, m, 32));
  return v;
}
__device__ __forceinline__ float hsum16(float v) {
#pragma unroll
  for (int m = 1; m < 16; m <<= 1) v += __shfl_xor(v, m, 32);
  return v;
}

// ---------------- fp32 -> bf16 cast (weights) ----------------
__global__ void cast_bf16_kernel(const float* __restrict__ in,
                                 unsigned short* __restrict__ out, int n) {
  for (int i = blockIdx.x * blockDim.x + threadIdx.x; i < n; i += gridDim.x * blockDim.x)
    out[i] = f2bf(in[i]);
}

// ---------------- LayerNorm: fp32 in -> bf16 out ----------------
__global__ __launch_bounds__(256) void ln_kernel(const float* __restrict__ x,
                                                 const float* __restrict__ g,
                                                 const float* __restrict__ b,
                                                 unsigned short* __restrict__ out) {
  __shared__ float rs[256], rq[256];
  const int row = blockIdx.x;
  const float* xr = x + (size_t)row * Hd;
  float s = 0.f, s2 = 0.f;
  for (int i = threadIdx.x; i < Hd; i += 256) { float v = xr[i]; s += v; s2 += v * v; }
  rs[threadIdx.x] = s; rq[threadIdx.x] = s2;
  __syncthreads();
  for (int o = 128; o > 0; o >>= 1) {
    if (threadIdx.x < o) { rs[threadIdx.x] += rs[threadIdx.x + o]; rq[threadIdx.x] += rq[threadIdx.x + o]; }
    __syncthreads();
  }
  const float mu  = rs[0] * (1.f / Hd);
  const float var = rq[0] * (1.f / Hd) - mu * mu;
  const float ri  = rsqrtf(var + 1e-6f);
  for (int i = threadIdx.x; i < Hd; i += 256)
    out[(size_t)row * Hd + i] = f2bf((xr[i] - mu) * ri * g[i] + b[i]);
}

// ---------------- Tiled bf16 GEMM: C[M,N] = A[M,K] * B[N,K]^T + epilogue ----
// 256 threads = 8 waves; block tile 128x128; wave tile 32x64 (2x4 wmma tiles);
// K-step 32; double-buffered LDS fed by the Tensor Data Mover (wave 0 issues,
// TENSORcnt + barrier synchronize) so no VGPRs are burned on staging.
#define BM 128
#define BN 128
#define BK 32
__global__ __launch_bounds__(256) void gemm_kernel(
    const unsigned short* __restrict__ A,   // [M,K] bf16
    const unsigned short* __restrict__ Bw,  // [N,K] bf16 (i.e. W, row-major)
    const float* __restrict__ bias,         // [N] or null
    const float* __restrict__ resid,        // [M,N] fp32 or null
    float* __restrict__ Cf,                 // [M,N] fp32 or null
    unsigned short* __restrict__ Cb,        // [M,N] bf16 or null
    int M, int N, int K, int do_gelu) {
  __shared__ unsigned short As[2][BM * BK];
  __shared__ unsigned short Bs[2][BN * BK];
  const int tid  = threadIdx.x;
  const int lane = tid & 31;
  const int wave = tid >> 5;
  const int wm   = wave & 3;     // 4 waves along M (32 rows each)
  const int wn   = wave >> 2;    // 2 waves along N (64 cols each)
  const int m0   = blockIdx.y * BM;
  const int n0   = blockIdx.x * BN;

  v8f acc[2][4];
#pragma unroll
  for (int i = 0; i < 2; ++i)
#pragma unroll
    for (int j = 0; j < 4; ++j) acc[i][j] = (v8f)(0.f);

  const int nk = K / BK;

#if USE_TDM
  if (wave == 0) {
    tdm_load_2d(A  + (size_t)m0 * K, ldsoff(As[0]), BK, BM, K);
    tdm_load_2d(Bw + (size_t)n0 * K, ldsoff(Bs[0]), BK, BN, K);
    __builtin_amdgcn_s_wait_tensorcnt(0);
  }
  __syncthreads();
  for (int kt = 0; kt < nk; ++kt) {
    const int buf = kt & 1;
    if (wave == 0 && kt + 1 < nk) {           // DMA next slice during WMMA
      tdm_load_2d(A  + (size_t)m0 * K + (kt + 1) * BK, ldsoff(As[buf ^ 1]), BK, BM, K);
      tdm_load_2d(Bw + (size_t)n0 * K + (kt + 1) * BK, ldsoff(Bs[buf ^ 1]), BK, BN, K);
    }
    Frag af[2], bf[4];
#pragma unroll
    for (int mt = 0; mt < 2; ++mt)
      load_afrag(af[mt], &As[buf][(wm * 32 + mt * 16) * BK], BK);
#pragma unroll
    for (int nt = 0; nt < 4; ++nt)
      load_bfrag(bf[nt], &Bs[buf][(wn * 64 + nt * 16) * BK], BK);
#pragma unroll
    for (int mt = 0; mt < 2; ++mt)
#pragma unroll
      for (int nt = 0; nt < 4; ++nt)
        acc[mt][nt] = wmma_bf16(af[mt], bf[nt], acc[mt][nt]);
    if (wave == 0 && kt + 1 < nk) __builtin_amdgcn_s_wait_tensorcnt(0);
    __syncthreads();
  }
#else
  uint4 ra[2], rb[2];
  auto fetch = [&](int kt) {
    const int k0 = kt * BK;
#pragma unroll
    for (int i = 0; i < 2; ++i) {
      const int s = tid * 2 + i;
      const int row = s >> 2, seg = s & 3;
      ra[i] = *(const uint4*)(A  + (size_t)(m0 + row) * K + k0 + seg * 8);
      rb[i] = *(const uint4*)(Bw + (size_t)(n0 + row) * K + k0 + seg * 8);
    }
  };
  auto stage = [&](int buf) {
#pragma unroll
    for (int i = 0; i < 2; ++i) {
      const int s = tid * 2 + i;
      ((uint4*)As[buf])[s] = ra[i];
      ((uint4*)Bs[buf])[s] = rb[i];
    }
  };
  fetch(0); stage(0);
  __syncthreads();
  for (int kt = 0; kt < nk; ++kt) {
    const int buf = kt & 1;
    if (kt + 1 < nk) fetch(kt + 1);
    Frag af[2], bf[4];
#pragma unroll
    for (int mt = 0; mt < 2; ++mt)
      load_afrag(af[mt], &As[buf][(wm * 32 + mt * 16) * BK], BK);
#pragma unroll
    for (int nt = 0; nt < 4; ++nt)
      load_bfrag(bf[nt], &Bs[buf][(wn * 64 + nt * 16) * BK], BK);
#pragma unroll
    for (int mt = 0; mt < 2; ++mt)
#pragma unroll
      for (int nt = 0; nt < 4; ++nt)
        acc[mt][nt] = wmma_bf16(af[mt], bf[nt], acc[mt][nt]);
    if (kt + 1 < nk) stage(buf ^ 1);
    __syncthreads();
  }
#endif

  // epilogue: C/D layout -> lane&15 = col, lane half picks rows 0-7 / 8-15
  const int colL  = lane & 15;
  const int rhalf = (lane < 16) ? 0 : 8;
#pragma unroll
  for (int mt = 0; mt < 2; ++mt)
#pragma unroll
    for (int nt = 0; nt < 4; ++nt) {
      const int n = n0 + wn * 64 + nt * 16 + colL;
#pragma unroll
      for (int v = 0; v < 8; ++v) {
        const int m = m0 + wm * 32 + mt * 16 + rhalf + v;
        float val = acc[mt][nt][v];
        if (bias) val += bias[n];
        if (do_gelu) val = 0.5f * val * (1.f + erff(val * 0.70710678118654752f));
        if (resid) val += resid[(size_t)m * N + n];
        if (Cf) Cf[(size_t)m * N + n] = val;
        if (Cb) Cb[(size_t)m * N + n] = f2bf(val);
      }
    }
}

// ---------------- Flash attention (causal), bf16 WMMA, fp32 softmax --------
// grid = (S/128, NH, B); block = 256 (8 waves); each wave owns 16 q rows.
// Q tile and K chunks arrive via TDM; V is transposed manually into LDS.
__global__ __launch_bounds__(256) void attn_kernel(
    const unsigned short* __restrict__ qkv,  // [B*S, 3H] bf16
    unsigned short* __restrict__ ctx) {      // [B*S, H]  bf16
  __shared__ unsigned short Qs[128 * 64];    // [q][d]
  __shared__ unsigned short Ks[64 * 64];     // [key][d]
  __shared__ unsigned short Vt[64 * 64];     // [d][key]  (transposed)
  __shared__ unsigned short Ps[128 * 64];    // [q][key]  bf16 probs

  const int tid  = threadIdx.x;
  const int lane = tid & 31;
  const int wave = tid >> 5;
  const int qb   = blockIdx.x;
  const int head = blockIdx.y;
  const int b    = blockIdx.z;
  const size_t qkv_base = (size_t)b * Sq * H3;

#if USE_TDM
  if (wave == 0)
    tdm_load_2d(qkv + qkv_base + (size_t)(qb * 128) * H3 + head * HDd,
                ldsoff(Qs), 64, 128, H3);
#else
#pragma unroll
  for (int i = 0; i < 4; ++i) {
    const int s = tid * 4 + i;
    const int row = s >> 3, seg = s & 7;
    ((uint4*)Qs)[s] = *(const uint4*)(qkv + qkv_base +
        (size_t)(qb * 128 + row) * H3 + head * HDd + seg * 8);
  }
#endif

  float mrow[8], lrow[8];
#pragma unroll
  for (int v = 0; v < 8; ++v) { mrow[v] = -1e30f; lrow[v] = 0.f; }
  v8f oacc[4];
#pragma unroll
  for (int nt = 0; nt < 4; ++nt) oacc[nt] = (v8f)(0.f);

  const int colL  = lane & 15;
  const int rhalf = (lane < 16) ? 0 : 8;
  const int nchunks = 2 * qb + 2;

  for (int kb = 0; kb < nchunks; ++kb) {
    // stage K chunk (TDM) and V chunk transposed (manual)
#if USE_TDM
    if (wave == 0)
      tdm_load_2d(qkv + qkv_base + (size_t)(kb * 64) * H3 + Hd + head * HDd,
                  ldsoff(Ks), 64, 64, H3);
#pragma unroll
    for (int i = 0; i < 2; ++i) {
      const int s = tid * 2 + i;
      const int row = s >> 3, seg = s & 7;
      U4 u; u.q = *(const uint4*)(qkv + qkv_base +
          (size_t)(kb * 64 + row) * H3 + 2 * Hd + head * HDd + seg * 8);
#pragma unroll
      for (int j = 0; j < 8; ++j) Vt[(seg * 8 + j) * 64 + row] = u.h[j];
    }
    if (wave == 0) __builtin_amdgcn_s_wait_tensorcnt(0);
#else
#pragma unroll
    for (int i = 0; i < 2; ++i) {
      const int s = tid * 2 + i;
      const int row = s >> 3, seg = s & 7;
      const size_t krow = qkv_base + (size_t)(kb * 64 + row) * H3 + head * HDd;
      ((uint4*)Ks)[s] = *(const uint4*)(qkv + krow + Hd + seg * 8);
      U4 u; u.q = *(const uint4*)(qkv + krow + 2 * Hd + seg * 8);
#pragma unroll
      for (int j = 0; j < 8; ++j) Vt[(seg * 8 + j) * 64 + row] = u.h[j];
    }
#endif
    __syncthreads();

    // scores: S = Q * K^T  (M=16, N=64 keys, inner HD=64 -> 2 K-steps)
    Frag qa[2];
    load_afrag(qa[0], Qs + (wave * 16) * 64, 64);
    load_afrag(qa[1], Qs + (wave * 16) * 64 + 32, 64);
    v8f st[4];
#pragma unroll
    for (int nt = 0; nt < 4; ++nt) {
      v8f sacc = (v8f)(0.f);
      Frag kf;
      load_bfrag(kf, Ks + nt * 16 * 64, 64);      sacc = wmma_bf16(qa[0], kf, sacc);
      load_bfrag(kf, Ks + nt * 16 * 64 + 32, 64); sacc = wmma_bf16(qa[1], kf, sacc);
      st[nt] = sacc;
    }

    // scale + causal mask + online softmax
    float rm[8];
#pragma unroll
    for (int v = 0; v < 8; ++v) rm[v] = -1e30f;
#pragma unroll
    for (int nt = 0; nt < 4; ++nt)
#pragma unroll
      for (int v = 0; v < 8; ++v) {
        float sv = st[nt][v] * 0.125f;            // HD^-0.5
        const int c  = kb * 64 + nt * 16 + colL;
        const int rg = qb * 128 + wave * 16 + rhalf + v;
        if (c > rg) sv = -1e30f;
        st[nt][v] = sv;
        rm[v] = fmaxf(rm[v], sv);
      }
    float sc[8], ls[8];
#pragma unroll
    for (int v = 0; v < 8; ++v) {
      rm[v] = hmax16(rm[v]);
      const float mn = fmaxf(mrow[v], rm[v]);
      sc[v] = __expf(mrow[v] - mn);
      mrow[v] = mn;
      ls[v] = 0.f;
    }
#pragma unroll
    for (int nt = 0; nt < 4; ++nt)
#pragma unroll
      for (int v = 0; v < 8; ++v) {
        const float p = __expf(st[nt][v] - mrow[v]);
        ls[v] += p;
        Ps[(wave * 16 + rhalf + v) * 64 + nt * 16 + colL] = f2bf(p);
      }
#pragma unroll
    for (int v = 0; v < 8; ++v) {
      ls[v] = hsum16(ls[v]);
      lrow[v] = lrow[v] * sc[v] + ls[v];
    }
#pragma unroll
    for (int nt = 0; nt < 4; ++nt)
#pragma unroll
      for (int v = 0; v < 8; ++v) oacc[nt][v] *= sc[v];

    // ctx += P * V  (M=16, N=HD=64, inner 64 keys -> 2 K-steps)
#pragma unroll
    for (int ks = 0; ks < 2; ++ks) {
      Frag pa;
      load_afrag(pa, Ps + (wave * 16) * 64 + ks * 32, 64);
#pragma unroll
      for (int nt = 0; nt < 4; ++nt) {
        Frag vf;
        load_bfrag(vf, Vt + nt * 16 * 64 + ks * 32, 64);
        oacc[nt] = wmma_bf16(pa, vf, oacc[nt]);
      }
    }
    __syncthreads();   // before next chunk overwrites Ks/Vt
  }

  // normalize + store ctx (already in [B,S,H] head-interleaved layout)
#pragma unroll
  for (int nt = 0; nt < 4; ++nt)
#pragma unroll
    for (int v = 0; v < 8; ++v) {
      const int rg = qb * 128 + wave * 16 + rhalf + v;
      const float o = oacc[nt][v] / lrow[v];
      ctx[((size_t)b * Sq + rg) * Hd + head * HDd + nt * 16 + colL] = f2bf(o);
    }
}

// ======================== launcher =========================================
extern "C" void kernel_launch(void* const* d_in, const int* in_sizes, int n_in,
                              void* d_out, int out_size, void* d_ws, size_t ws_size,
                              hipStream_t stream) {
  const float* x      = (const float*)d_in[0];
  // d_in[1] = attention_mask (causal triu) — realized analytically, unused
  const float* ln1_g  = (const float*)d_in[2];
  const float* ln1_b  = (const float*)d_in[3];
  const float* ln2_g  = (const float*)d_in[4];
  const float* ln2_b  = (const float*)d_in[5];
  const float* qkv_w  = (const float*)d_in[6];
  const float* out_w  = (const float*)d_in[7];
  const float* fc1_w  = (const float*)d_in[8];
  const float* fc1_b  = (const float*)d_in[9];
  const float* fc2_w  = (const float*)d_in[10];
  const float* fc2_b  = (const float*)d_in[11];
  float* out = (float*)d_out;

  char* base = (char*)d_ws;
  size_t off = 0;
  auto take = [&](size_t bytes) {
    char* p = base + off;
    off += (bytes + 255) & ~(size_t)255;
    return p;
  };
  unsigned short* wq  = (unsigned short*)take((size_t)H3 * Hd * 2);
  unsigned short* wo  = (unsigned short*)take((size_t)Hd * Hd * 2);
  unsigned short* w1  = (unsigned short*)take((size_t)DFFn * Hd * 2);
  unsigned short* w2  = (unsigned short*)take((size_t)Hd * DFFn * 2);
  unsigned short* h1  = (unsigned short*)take((size_t)Mrows * Hd * 2);
  unsigned short* qkv = (unsigned short*)take((size_t)Mrows * H3 * 2);
  unsigned short* ctx = (unsigned short*)take((size_t)Mrows * Hd * 2);
  float*          x1  = (float*)take((size_t)Mrows * Hd * 4);
  unsigned short* h2  = (unsigned short*)take((size_t)Mrows * Hd * 2);
  unsigned short* act = (unsigned short*)take((size_t)Mrows * DFFn * 2);
  (void)ws_size; (void)in_sizes; (void)n_in; (void)out_size;

  cast_bf16_kernel<<<4096, 256, 0, stream>>>(qkv_w, wq, H3 * Hd);
  cast_bf16_kernel<<<4096, 256, 0, stream>>>(out_w, wo, Hd * Hd);
  cast_bf16_kernel<<<4096, 256, 0, stream>>>(fc1_w, w1, DFFn * Hd);
  cast_bf16_kernel<<<4096, 256, 0, stream>>>(fc2_w, w2, Hd * DFFn);

  ln_kernel<<<Mrows, 256, 0, stream>>>(x, ln1_g, ln1_b, h1);
  gemm_kernel<<<dim3(H3 / BN, Mrows / BM), 256, 0, stream>>>(
      h1, wq, nullptr, nullptr, nullptr, qkv, Mrows, H3, Hd, 0);
  attn_kernel<<<dim3(Sq / 128, NHn, Bsz), 256, 0, stream>>>(qkv, ctx);
  gemm_kernel<<<dim3(Hd / BN, Mrows / BM), 256, 0, stream>>>(
      ctx, wo, nullptr, x, x1, nullptr, Mrows, Hd, Hd, 0);
  ln_kernel<<<Mrows, 256, 0, stream>>>(x1, ln2_g, ln2_b, h2);
  gemm_kernel<<<dim3(DFFn / BN, Mrows / BM), 256, 0, stream>>>(
      h2, w1, fc1_b, nullptr, nullptr, act, Mrows, DFFn, Hd, 1);
  gemm_kernel<<<dim3(Hd / BN, Mrows / BM), 256, 0, stream>>>(
      act, w2, fc2_b, x1, out, nullptr, Mrows, Hd, DFFn, 0);
}
